// simpleKT_369367188022
// MI455X (gfx1250) — matrix operand order repaired and checked by reading, not compiled, not run on hardware
//
#include <hip/hip_runtime.h>
#include <hip/hip_bf16.h>
#include <math.h>

#define B_    32
#define L_    512
#define D_    512
#define H_    8
#define NB_   2
#define DFF_  256
#define FC1_  512
#define FC2_  256
#define NPID_ 10000
#define M_    (B_ * L_)          // 16384 rows
#define ATT_WAVES 2

typedef __bf16 bf16;
typedef __attribute__((ext_vector_type(16))) __bf16 v16bf;
typedef __attribute__((ext_vector_type(8)))  float  v8f;

#define WMMA_BF16(a, b, c) \
  __builtin_amdgcn_wmma_f32_16x16x32_bf16(false, (a), false, (b), (short)0, (c), false, false)

// ---------- helpers ----------
__device__ __forceinline__ bf16 f2bf(float f) {
  union { float f; unsigned u; } a; a.f = f;
  unsigned r = a.u + 0x7FFFu + ((a.u >> 16) & 1u);     // round-to-nearest-even
  unsigned short h = (unsigned short)(r >> 16);
  union { unsigned short u; bf16 b; } c; c.u = h; return c.b;
}

union FragU { v16bf v; struct { uint4 lo, hi; } q; };

// A-operand (16x32 bf16): lane holds row (lane&15); caller passes row_ptr + (lane>>4)*8.
// elements 0..7 <- [0..7], elements 8..15 <- [16..23]  (two b128 loads)
__device__ __forceinline__ v16bf load_frag_a(const bf16* p) {
  FragU u;
  u.q.lo = *(const uint4*)(p);
  u.q.hi = *(const uint4*)(p + 16);
  return u.v;
}

// B-operand (32x16 bf16): lane holds column (lane&15); caller passes col_row_ptr + (lane>>4)*8.
// elements 0..7 <- [0..7], elements 8..15 <- [16..23]  (two b128 loads)
__device__ __forceinline__ v16bf load_frag_b(const bf16* p) {
  FragU u;
  u.q.lo = *(const uint4*)(p);
  u.q.hi = *(const uint4*)(p + 16);
  return u.v;
}

// A-operand built from f32 data (LDS softmax probs), converting to bf16 in VALU
__device__ __forceinline__ v16bf load_frag_a_f32(const float* p) {
  v16bf r;
#pragma unroll
  for (int j = 0; j < 8; j++) { r[j] = f2bf(p[j]); r[8 + j] = f2bf(p[16 + j]); }
  return r;
}

// ---------- embedding + positional encoding ----------
__global__ void k_embed(const int* __restrict__ prob, const int* __restrict__ corr,
                        const float* __restrict__ q_embed, const float* __restrict__ qa_embed,
                        const float* __restrict__ q_diff, const float* __restrict__ dparam,
                        float* __restrict__ x, bf16* __restrict__ xbf,
                        float* __restrict__ qe_out, bf16* __restrict__ ybf) {
  int idx = blockIdx.x * blockDim.x + threadIdx.x;
  if (idx >= M_ * D_) return;
  int d  = idx & (D_ - 1);
  int bl = idx >> 9;
  int l  = bl & (L_ - 1);
  int qd = prob[bl] + 1;
  int qa = qd + corr[bl] * NPID_;
  size_t ed = (size_t)qd * D_ + d;
  float qe = q_embed[ed] + dparam[ed] * q_diff[ed];
  int i2 = d & ~1;
  float ang = (float)l * expf((float)i2 * (-9.2103403720f / (float)D_));
  float pe = (d & 1) ? cosf(ang) : sinf(ang);
  float xv = qe + pe;
  qe_out[idx] = qe;
  x[idx] = xv;
  xbf[idx] = f2bf(xv);
  ybf[idx] = f2bf(qa_embed[(size_t)qa * D_ + d] + pe);
}

// ---------- weight transpose + convert: W[K,N] f32 -> Wt[N,K] bf16 ----------
__global__ void k_transw(const float* __restrict__ W, bf16* __restrict__ Wt, int K, int N) {
  int idx = blockIdx.x * blockDim.x + threadIdx.x;
  if (idx >= K * N) return;
  int k = idx / N, n = idx - k * N;
  Wt[(size_t)n * K + k] = f2bf(W[idx]);
}

// ---------- concat [x, qe] -> bf16 [M, 2D], 4 elements/thread ----------
__global__ void k_concat(const float* __restrict__ x, const float* __restrict__ qe,
                         bf16* __restrict__ A) {
  int idx = blockIdx.x * blockDim.x + threadIdx.x;
  if (idx >= M_ * 2 * D_ / 4) return;
  int g = idx & (2 * D_ / 4 - 1);
  int m = idx >> 8;
  int c = g << 2;
  const float* src = (c < D_) ? (x + (size_t)m * D_ + c) : (qe + (size_t)m * D_ + (c - D_));
  float4 f = *(const float4*)src;
  union { uint2 u; bf16 e[4]; } pk;
  pk.e[0] = f2bf(f.x); pk.e[1] = f2bf(f.y); pk.e[2] = f2bf(f.z); pk.e[3] = f2bf(f.w);
  *(uint2*)(A + (size_t)m * 2 * D_ + c) = pk.u;
}

// ---------- WMMA GEMM: one wave -> 32x64 tile, ping-pong pipelined (K % 64 == 0) ----
// mode 0: C f32 row-major; mode 1: Cb bf16 row-major; mode 2: Cb bf16 in [b,d,L] (V-transpose)
__global__ void k_gemm(const bf16* __restrict__ A, const bf16* __restrict__ Wt,
                       const float* __restrict__ bias, float* __restrict__ C,
                       bf16* __restrict__ Cb, int Mrows, int N, int K, int relu, int mode) {
  int gid  = blockIdx.x * blockDim.x + threadIdx.x;
  int wave = gid >> 5;
  int lane = threadIdx.x & 31;
  int nstr = N >> 6;
  int m0 = (wave / nstr) << 5;     // 32 output rows
  int n0 = (wave % nstr) << 6;     // 64 output cols
  if (m0 >= Mrows) return;
  int mrow = lane & 15;
  int hi   = lane >> 4;

  size_t wstep = (size_t)16 * K;
  const bf16* pa0 = A + (size_t)(m0 + mrow) * K + hi * 8;
  const bf16* pa1 = pa0 + (size_t)16 * K;
  const bf16* pw0 = Wt + (size_t)(n0 + mrow) * K + hi * 8;
  const bf16* pw1 = pw0 + wstep;
  const bf16* pw2 = pw1 + wstep;
  const bf16* pw3 = pw2 + wstep;

  v8f z = {0.f,0.f,0.f,0.f,0.f,0.f,0.f,0.f};
  v8f acc[2][4];
#pragma unroll
  for (int mi = 0; mi < 2; mi++)
#pragma unroll
    for (int t = 0; t < 4; t++) acc[mi][t] = z;

  // set C = fragments at k; set N = fragments at k+32  (no register copies)
  v16bf aC0 = load_frag_a(pa0), aC1 = load_frag_a(pa1);
  v16bf bC0 = load_frag_b(pw0), bC1 = load_frag_b(pw1);
  v16bf bC2 = load_frag_b(pw2), bC3 = load_frag_b(pw3);

  for (int k = 0; k < K - 64; k += 64) {
    v16bf aN0 = load_frag_a(pa0 + 32), aN1 = load_frag_a(pa1 + 32);
    v16bf bN0 = load_frag_b(pw0 + 32), bN1 = load_frag_b(pw1 + 32);
    v16bf bN2 = load_frag_b(pw2 + 32), bN3 = load_frag_b(pw3 + 32);
    acc[0][0] = WMMA_BF16(aC0, bC0, acc[0][0]);
    acc[0][1] = WMMA_BF16(aC0, bC1, acc[0][1]);
    acc[0][2] = WMMA_BF16(aC0, bC2, acc[0][2]);
    acc[0][3] = WMMA_BF16(aC0, bC3, acc[0][3]);
    acc[1][0] = WMMA_BF16(aC1, bC0, acc[1][0]);
    acc[1][1] = WMMA_BF16(aC1, bC1, acc[1][1]);
    acc[1][2] = WMMA_BF16(aC1, bC2, acc[1][2]);
    acc[1][3] = WMMA_BF16(aC1, bC3, acc[1][3]);
    pa0 += 64; pa1 += 64; pw0 += 64; pw1 += 64; pw2 += 64; pw3 += 64;
    aC0 = load_frag_a(pa0); aC1 = load_frag_a(pa1);
    bC0 = load_frag_b(pw0); bC1 = load_frag_b(pw1);
    bC2 = load_frag_b(pw2); bC3 = load_frag_b(pw3);
    acc[0][0] = WMMA_BF16(aN0, bN0, acc[0][0]);
    acc[0][1] = WMMA_BF16(aN0, bN1, acc[0][1]);
    acc[0][2] = WMMA_BF16(aN0, bN2, acc[0][2]);
    acc[0][3] = WMMA_BF16(aN0, bN3, acc[0][3]);
    acc[1][0] = WMMA_BF16(aN1, bN0, acc[1][0]);
    acc[1][1] = WMMA_BF16(aN1, bN1, acc[1][1]);
    acc[1][2] = WMMA_BF16(aN1, bN2, acc[1][2]);
    acc[1][3] = WMMA_BF16(aN1, bN3, acc[1][3]);
  }
  {
    v16bf aN0 = load_frag_a(pa0 + 32), aN1 = load_frag_a(pa1 + 32);
    v16bf bN0 = load_frag_b(pw0 + 32), bN1 = load_frag_b(pw1 + 32);
    v16bf bN2 = load_frag_b(pw2 + 32), bN3 = load_frag_b(pw3 + 32);
    acc[0][0] = WMMA_BF16(aC0, bC0, acc[0][0]);
    acc[0][1] = WMMA_BF16(aC0, bC1, acc[0][1]);
    acc[0][2] = WMMA_BF16(aC0, bC2, acc[0][2]);
    acc[0][3] = WMMA_BF16(aC0, bC3, acc[0][3]);
    acc[1][0] = WMMA_BF16(aC1, bC0, acc[1][0]);
    acc[1][1] = WMMA_BF16(aC1, bC1, acc[1][1]);
    acc[1][2] = WMMA_BF16(aC1, bC2, acc[1][2]);
    acc[1][3] = WMMA_BF16(aC1, bC3, acc[1][3]);
    acc[0][0] = WMMA_BF16(aN0, bN0, acc[0][0]);
    acc[0][1] = WMMA_BF16(aN0, bN1, acc[0][1]);
    acc[0][2] = WMMA_BF16(aN0, bN2, acc[0][2]);
    acc[0][3] = WMMA_BF16(aN0, bN3, acc[0][3]);
    acc[1][0] = WMMA_BF16(aN1, bN0, acc[1][0]);
    acc[1][1] = WMMA_BF16(aN1, bN1, acc[1][1]);
    acc[1][2] = WMMA_BF16(aN1, bN2, acc[1][2]);
    acc[1][3] = WMMA_BF16(aN1, bN3, acc[1][3]);
  }

  // epilogue
  if (mode == 2) {                       // V-transpose store: vt[(b*D+col)*L + l], N == D_
    int bidx = m0 >> 9;                  // 32-row tile stays inside one batch element
#pragma unroll
    for (int mi = 0; mi < 2; mi++) {
      int l0 = ((m0 + mi * 16) & (L_ - 1)) + 8 * hi;
#pragma unroll
      for (int t = 0; t < 4; t++) {
        int col = n0 + (t << 4) + mrow;
        float bz = bias ? bias[col] : 0.f;
        union { uint4 q; bf16 e[8]; } pk;
#pragma unroll
        for (int r = 0; r < 8; r++) pk.e[r] = f2bf(acc[mi][t][r] + bz);
        *(uint4*)(Cb + ((size_t)bidx * D_ + col) * L_ + l0) = pk.q;
      }
    }
  } else {
#pragma unroll
    for (int mi = 0; mi < 2; mi++)
#pragma unroll
      for (int t = 0; t < 4; t++) {
        int col = n0 + (t << 4) + mrow;
        float bz = bias ? bias[col] : 0.f;
#pragma unroll
        for (int r = 0; r < 8; r++) {
          int row = m0 + mi * 16 + r + 8 * hi;
          float val = acc[mi][t][r] + bz;
          if (relu) val = fmaxf(val, 0.f);
          if (mode == 1) Cb[(size_t)row * N + col] = f2bf(val);
          else           C[(size_t)row * N + col] = val;
        }
      }
  }
}

// ---------- fused causal attention per (b, h, 16-query-row tile) ----------
// qk: [B,L,D] bf16 (q==k projection); vt: [(b*D+d)*L + l] bf16; o: [B,L,D] bf16
__global__ void k_attn(const bf16* __restrict__ qk, const bf16* __restrict__ vt,
                       bf16* __restrict__ o) {
  extern __shared__ float smem[];
  int wib  = threadIdx.x >> 5;
  int lane = threadIdx.x & 31;
  int gw   = blockIdx.x * ATT_WAVES + wib;       // 0 .. B*H*(L/16)-1 = 8191
  int qt = gw & 31;
  int h  = (gw >> 5) & 7;
  int b  = gw >> 8;
  int m0 = qt << 4;
  float* S = smem + wib * (16 * 512);
  int mrow = lane & 15, hi = lane >> 4;

  const bf16* qrow = qk + ((size_t)b * L_ + m0 + mrow) * D_ + h * 64 + hi * 8;
  v16bf a0 = load_frag_a(qrow);
  v16bf a1 = load_frag_a(qrow + 32);

  v8f z = {0.f,0.f,0.f,0.f,0.f,0.f,0.f,0.f};

  // scores S = (Q K^T) / 8, strict-causal; prefetch next key tile (clamped index)
  const bf16* kbase = qk + ((size_t)b * L_ + mrow) * D_ + h * 64 + hi * 8;
  v16bf kb0 = load_frag_b(kbase);
  v16bf kb1 = load_frag_b(kbase + 32);
  for (int n0 = 0; n0 <= m0; n0 += 16) {
    int np = (n0 < m0) ? n0 + 16 : n0;
    const bf16* kr = kbase + (size_t)np * D_;
    v16bf nb0 = load_frag_b(kr);
    v16bf nb1 = load_frag_b(kr + 32);
    v8f acc = z;
    acc = WMMA_BF16(a0, kb0, acc);
    acc = WMMA_BF16(a1, kb1, acc);
#pragma unroll
    for (int r = 0; r < 8; r++) {
      int rl = r + 8 * hi;
      float sv = acc[r] * 0.125f;
      if (n0 + mrow >= m0 + rl) sv = -1e30f;     // attend only to strictly-past keys
      S[rl * 512 + n0 + mrow] = sv;
    }
    kb0 = nb0; kb1 = nb1;
  }

  int ncols  = m0 + 16;
  int ncols2 = (ncols + 31) & ~31;
  // row-wise softmax in LDS (P stored back as f32); row 0 zeroed (zero_pad)
  for (int r = 0; r < 16; r++) {
    float* Sr = S + r * 512;
    int grow = m0 + r;
    if (grow == 0) {
      for (int c = lane; c < ncols2; c += 32) Sr[c] = 0.f;
      continue;
    }
    float mx = -3.0e38f;
    for (int c = lane; c < ncols; c += 32) mx = fmaxf(mx, Sr[c]);
#pragma unroll
    for (int o2 = 16; o2 >= 1; o2 >>= 1) mx = fmaxf(mx, __shfl_xor(mx, o2, 32));
    float sum = 0.f;
    for (int c = lane; c < ncols; c += 32) { float e = __expf(Sr[c] - mx); Sr[c] = e; sum += e; }
#pragma unroll
    for (int o2 = 16; o2 >= 1; o2 >>= 1) sum += __shfl_xor(sum, o2, 32);
    float inv = 1.f / sum;
    for (int c = lane; c < ncols; c += 32) Sr[c] *= inv;
    for (int c = ncols + lane; c < ncols2; c += 32) Sr[c] = 0.f;   // zero tail for K-step 32
  }

  // O = P @ V  (dk = 64 -> 4 accumulators), prefetch next V tiles
  v8f accO[4];
#pragma unroll
  for (int t = 0; t < 4; t++) accO[t] = z;
  const bf16* vbase = vt + ((size_t)b * D_ + h * 64 + mrow) * L_ + hi * 8;
  v16bf vb_cur[4];
#pragma unroll
  for (int t = 0; t < 4; t++) vb_cur[t] = load_frag_b(vbase + (size_t)(t << 4) * L_);
  for (int kc = 0; kc < ncols; kc += 32) {
    int kn = (kc + 32 < ncols) ? kc + 32 : kc;
    v16bf vb_nxt[4];
#pragma unroll
    for (int t = 0; t < 4; t++) vb_nxt[t] = load_frag_b(vbase + (size_t)(t << 4) * L_ + kn);
    v16bf pa = load_frag_a_f32(S + mrow * 512 + kc + hi * 8);
#pragma unroll
    for (int t = 0; t < 4; t++) accO[t] = WMMA_BF16(pa, vb_cur[t], accO[t]);
#pragma unroll
    for (int t = 0; t < 4; t++) vb_cur[t] = vb_nxt[t];
  }
#pragma unroll
  for (int t = 0; t < 4; t++)
#pragma unroll
    for (int r = 0; r < 8; r++)
      o[((size_t)b * L_ + m0 + r + 8 * hi) * D_ + h * 64 + t * 16 + mrow] = f2bf(accO[t][r]);
}

// ---------- residual add + LayerNorm (one wave per row); bf16 copy ----------
__global__ void k_addln(float* __restrict__ x, const float* __restrict__ t,
                        const float* __restrict__ s, const float* __restrict__ bb,
                        bf16* __restrict__ xb) {
  int wid  = (blockIdx.x * blockDim.x + threadIdx.x) >> 5;
  int lane = threadIdx.x & 31;
  if (wid >= M_) return;
  size_t base = (size_t)wid * D_;
  float v[16];
  float sum = 0.f;
#pragma unroll
  for (int i = 0; i < 16; i++) {
    int c = lane + (i << 5);
    float zz = x[base + c] + t[base + c];
    v[i] = zz; sum += zz;
  }
#pragma unroll
  for (int o2 = 16; o2 >= 1; o2 >>= 1) sum += __shfl_xor(sum, o2, 32);
  float mean = sum * (1.f / (float)D_);
  float var = 0.f;
#pragma unroll
  for (int i = 0; i < 16; i++) { float d0 = v[i] - mean; var += d0 * d0; }
#pragma unroll
  for (int o2 = 16; o2 >= 1; o2 >>= 1) var += __shfl_xor(var, o2, 32);
  var *= (1.f / (float)D_);
  float inv = rsqrtf(var + 1e-5f);
#pragma unroll
  for (int i = 0; i < 16; i++) {
    int c = lane + (i << 5);
    float val = (v[i] - mean) * inv * s[c] + bb[c];
    x[base + c] = val;
    xb[base + c] = f2bf(val);
  }
}

// ---------- final: one wave per row; logits = h2 @ w3 + b3; sigmoid; skip l==0 ----------
__global__ void k_head(const float* __restrict__ h2, const float* __restrict__ w3,
                       const float* __restrict__ b3, float* __restrict__ out) {
  int wid  = (blockIdx.x * blockDim.x + threadIdx.x) >> 5;
  int lane = threadIdx.x & 31;
  if (wid >= M_) return;
  const float* hr = h2 + (size_t)wid * FC2_;
  float acc = 0.f;
#pragma unroll
  for (int i = 0; i < FC2_ / 32; i++) {
    int k = lane + (i << 5);
    acc += hr[k] * w3[k];
  }
#pragma unroll
  for (int o2 = 16; o2 >= 1; o2 >>= 1) acc += __shfl_xor(acc, o2, 32);
  if (lane == 0) {
    int l = wid & (L_ - 1);
    int b = wid >> 9;
    float logit = acc + b3[0];
    if (l >= 1) out[(size_t)b * (L_ - 1) + (l - 1)] = 1.f / (1.f + expf(-logit));
  }
}

// ---------- host ----------
static inline dim3 grid1(size_t n) { return dim3((unsigned)((n + 255) / 256)); }

static void launch_gemm(const bf16* A, const bf16* Wt, const float* bias, float* C,
                        bf16* Cb, int M, int N, int K, int relu, int mode,
                        hipStream_t stream) {
  size_t waves = (size_t)(M / 32) * (N / 64);
  k_gemm<<<grid1(waves * 32), 256, 0, stream>>>(A, Wt, bias, C, Cb, M, N, K, relu, mode);
}

extern "C" void kernel_launch(void* const* d_in, const int* in_sizes, int n_in,
                              void* d_out, int out_size, void* d_ws, size_t ws_size,
                              hipStream_t stream) {
  (void)in_sizes; (void)n_in; (void)out_size; (void)ws_size;
  const int*   prob     = (const int*)d_in[0];
  const int*   corr     = (const int*)d_in[1];
  const float* q_embed  = (const float*)d_in[2];
  const float* qa_embed = (const float*)d_in[3];
  const float* q_diff   = (const float*)d_in[4];
  const float* dparam   = (const float*)d_in[5];
  const float* kw  = (const float*)d_in[6];
  const float* kb  = (const float*)d_in[7];
  const float* vw  = (const float*)d_in[8];
  const float* vb  = (const float*)d_in[9];
  const float* ow  = (const float*)d_in[10];
  const float* ob  = (const float*)d_in[11];
  const float* ln1s = (const float*)d_in[12];
  const float* ln1b = (const float*)d_in[13];
  const float* w1  = (const float*)d_in[14];
  const float* b1  = (const float*)d_in[15];
  const float* w2  = (const float*)d_in[16];
  const float* b2  = (const float*)d_in[17];
  const float* ln2s = (const float*)d_in[18];
  const float* ln2b = (const float*)d_in[19];
  const float* ow1 = (const float*)d_in[20];
  const float* ob1 = (const float*)d_in[21];
  const float* ow2 = (const float*)d_in[22];
  const float* ob2 = (const float*)d_in[23];
  const float* ow3 = (const float*)d_in[24];
  const float* ob3 = (const float*)d_in[25];
  float* out = (float*)d_out;

  // workspace bump allocator (~177 MB; working set fits MI455X's 192 MB L2)
  char* ws = (char*)d_ws;
  size_t off = 0;
  auto take = [&](size_t bytes) -> char* {
    char* p = ws + off; off += (bytes + 255) & ~(size_t)255; return p;
  };
  float* qe  = (float*)take((size_t)M_ * D_ * 4);
  float* x   = (float*)take((size_t)M_ * D_ * 4);
  bf16*  ybf = (bf16*) take((size_t)M_ * D_ * 2);
  bf16*  Abf = (bf16*) take((size_t)M_ * 2 * D_ * 2);  // bf16 GEMM input / attention O
  bf16*  Wt  = (bf16*) take((size_t)2 * D_ * FC1_ * 2);
  float* t0  = (float*)take((size_t)M_ * D_ * 4);
  bf16*  qkb = (bf16*) take((size_t)M_ * D_ * 2);      // qk bf16 / ffn1 out / head h1
  bf16*  vtb = (bf16*) take((size_t)M_ * D_ * 2);      // V transposed [b,d,L]

  k_embed<<<grid1((size_t)M_ * D_), 256, 0, stream>>>(prob, corr, q_embed, qa_embed,
                                                      q_diff, dparam, x, Abf, qe, ybf);

  for (int i = 0; i < NB_; i++) {
    // qk = x @ kw + kb  -> bf16 (q == k projection)
    k_transw<<<grid1((size_t)D_ * D_), 256, 0, stream>>>(kw + (size_t)i * D_ * D_, Wt, D_, D_);
    launch_gemm(Abf, Wt, kb + i * D_, nullptr, qkb, M_, D_, D_, 0, 1, stream);
    // v = y @ vw + vb  -> bf16 directly in [b,d,L] transposed layout
    k_transw<<<grid1((size_t)D_ * D_), 256, 0, stream>>>(vw + (size_t)i * D_ * D_, Wt, D_, D_);
    launch_gemm(ybf, Wt, vb + i * D_, nullptr, vtb, M_, D_, D_, 0, 2, stream);
    // fused causal attention -> O bf16 in Abf
    k_attn<<<dim3((B_ * H_ * (L_ / 16)) / ATT_WAVES), ATT_WAVES * 32,
             ATT_WAVES * 16 * 512 * 4, stream>>>(qkb, vtb, Abf);
    // x = LN(x + O @ ow + ob); bf16 copy of x -> Abf
    k_transw<<<grid1((size_t)D_ * D_), 256, 0, stream>>>(ow + (size_t)i * D_ * D_, Wt, D_, D_);
    launch_gemm(Abf, Wt, ob + i * D_, t0, nullptr, M_, D_, D_, 0, 0, stream);
    k_addln<<<grid1((size_t)M_ * 32), 256, 0, stream>>>(x, t0, ln1s + i * D_, ln1b + i * D_, Abf);
    // x = LN(x + relu(x@w1+b1)@w2 + b2); bf16 copy -> Abf
    k_transw<<<grid1((size_t)D_ * DFF_), 256, 0, stream>>>(w1 + (size_t)i * D_ * DFF_, Wt, D_, DFF_);
    launch_gemm(Abf, Wt, b1 + i * DFF_, nullptr, qkb, M_, DFF_, D_, 1, 1, stream);
    k_transw<<<grid1((size_t)DFF_ * D_), 256, 0, stream>>>(w2 + (size_t)i * DFF_ * D_, Wt, DFF_, D_);
    launch_gemm(qkb, Wt, b2 + i * D_, t0, nullptr, M_, D_, DFF_, 0, 0, stream);
    k_addln<<<grid1((size_t)M_ * 32), 256, 0, stream>>>(x, t0, ln2s + i * D_, ln2b + i * D_, Abf);
  }

  // head: h = relu(concat[x,qe] @ ow1 + ob1); h = relu(h @ ow2 + ob2); sigmoid(h @ ow3 + ob3)
  k_concat<<<grid1((size_t)M_ * 2 * D_ / 4), 256, 0, stream>>>(x, qe, Abf);
  k_transw<<<grid1((size_t)2 * D_ * FC1_), 256, 0, stream>>>(ow1, Wt, 2 * D_, FC1_);
  launch_gemm(Abf, Wt, ob1, nullptr, qkb, M_, FC1_, 2 * D_, 1, 1, stream);
  k_transw<<<grid1((size_t)FC1_ * FC2_), 256, 0, stream>>>(ow2, Wt, FC1_, FC2_);
  launch_gemm(qkb, Wt, ob2, t0, nullptr, M_, FC2_, FC1_, 1, 0, stream);
  k_head<<<grid1((size_t)M_ * 32), 256, 0, stream>>>(t0, ow3, ob3, out);
}
